// RNN_2860448219668
// MI455X (gfx1250) — matrix operand order repaired and checked by reading, not compiled, not run on hardware
//
#include <hip/hip_runtime.h>

// Elman RNN on gfx1250 (MI455X), f16 WMMA with f32 accumulate.
//   T=1024, B=128, DIM_IN=DIM_LAT=256
// Phase 1: xw = x @ Wi^T  -> written into the hs region of d_out
// Phase 2: serial scan h_t = tanh(xw_t + h_{t-1} @ Wh^T + bh), overwriting
//          xw_t in place with h_t; h_final copied from last hs slab.

typedef __attribute__((ext_vector_type(16))) _Float16 v16h;
typedef __attribute__((ext_vector_type(8)))  float    v8f;

#define T_STEPS 1024
#define BATCH   128
#define DIM     256

union HFrag {
    v16h v;
    uint4 q[2];
    _Float16 h[16];
};

// 16-value f16 fragment from two 16-byte-aligned chunks.
__device__ __forceinline__ v16h load_frag(const _Float16* p0, const _Float16* p1) {
    HFrag f;
    f.q[0] = *reinterpret_cast<const uint4*>(p0);
    f.q[1] = *reinterpret_cast<const uint4*>(p1);
    return f.v;
}

// ---------------------------------------------------------------------------
// Phase 1: xw[row, :] = x[row, :] @ Wi^T   (Wi row-major [n][k] == B layout)
// 256 blocks x 128 threads (4 waves); each wave owns a 16-row M-tile per iter.
// ---------------------------------------------------------------------------
__global__ __launch_bounds__(128) void xwi_gemm(const float* __restrict__ x,
                                                const float* __restrict__ Wi,
                                                float* __restrict__ xw) {
    __shared__ _Float16 wiLds[DIM * DIM];   // 128 KB, B-operand layout [n][k]

    const int tid = threadIdx.x;

    // Stage Wi (f32 -> f16) into LDS.
    {
        const float4* src = reinterpret_cast<const float4*>(Wi);
        for (int i = tid; i < (DIM * DIM) / 4; i += 128) {
            float4 f = src[i];
            _Float16* d = wiLds + i * 4;
            d[0] = (_Float16)f.x; d[1] = (_Float16)f.y;
            d[2] = (_Float16)f.z; d[3] = (_Float16)f.w;
        }
    }
    __syncthreads();

    const int wave = tid >> 5;
    const int lane = tid & 31;
    const int half = lane >> 4;
    const int mrow = lane & 15;

    const int waveGlobal = blockIdx.x * 4 + wave;
    const int nWaves     = gridDim.x * 4;
    const int nMTiles    = (T_STEPS * BATCH) / 16;   // 8192

    // Per-lane invariant base for the B fragments.
    const _Float16* const bBase = wiLds + mrow * DIM + half * 8;

    for (int mt = waveGlobal; mt < nMTiles; mt += nWaves) {
        // One base pointer per lane; all further addressing is constant-offset.
        const float* xr = x + (size_t)(mt * 16 + mrow) * DIM + half * 8;
        float* outp     = xw + (size_t)(mt * 16 + half * 8) * DIM + mrow;

        // A fragments: per lane K = kk*32 + half*8 + j (j<8), +16 for j>=8.
        v16h af[8];
#pragma unroll
        for (int kk = 0; kk < 8; ++kk) {
            const float4* p = reinterpret_cast<const float4*>(xr + kk * 32);
            const float4* q = reinterpret_cast<const float4*>(xr + kk * 32 + 16);
            float4 a0 = p[0], a1 = p[1], b0 = q[0], b1 = q[1];
            HFrag f;
            f.h[0]  = (_Float16)a0.x; f.h[1]  = (_Float16)a0.y;
            f.h[2]  = (_Float16)a0.z; f.h[3]  = (_Float16)a0.w;
            f.h[4]  = (_Float16)a1.x; f.h[5]  = (_Float16)a1.y;
            f.h[6]  = (_Float16)a1.z; f.h[7]  = (_Float16)a1.w;
            f.h[8]  = (_Float16)b0.x; f.h[9]  = (_Float16)b0.y;
            f.h[10] = (_Float16)b0.z; f.h[11] = (_Float16)b0.w;
            f.h[12] = (_Float16)b1.x; f.h[13] = (_Float16)b1.y;
            f.h[14] = (_Float16)b1.z; f.h[15] = (_Float16)b1.w;
            af[kk] = f.v;
        }

        const _Float16* bp = bBase;
#pragma unroll
        for (int nt = 0; nt < 16; ++nt) {
            // Preload all 8 B fragments so DS loads pipeline ahead of WMMAs.
            v16h bf[8];
#pragma unroll
            for (int kk = 0; kk < 8; ++kk)
                bf[kk] = load_frag(bp + kk * 32, bp + kk * 32 + 16);

            v8f c = {};
#pragma unroll
            for (int kk = 0; kk < 8; ++kk)
                c = __builtin_amdgcn_wmma_f32_16x16x32_f16(
                        false, af[kk], false, bf[kk], (short)0, c, false, false);

            // D layout: element e -> row (e + 8*half), col nt*16 + mrow.
#pragma unroll
            for (int e = 0; e < 8; ++e)
                outp[e * DIM + nt * 16] = c[e];

            bp += 16 * DIM;
        }
    }
}

// ---------------------------------------------------------------------------
// Phase 2: serial scan. 4 blocks x 128 threads (4 waves = 2 M-tiles x 2 N-halves).
// Wh (f16, [n][k]) resident in LDS; h ping-pongs through a double-buffered
// LDS staging tile (one barrier per step).
// ---------------------------------------------------------------------------
__global__ __launch_bounds__(128) void rnn_scan(const float* __restrict__ Wh,
                                                const float* __restrict__ bh,
                                                float* __restrict__ hFinal,
                                                float* __restrict__ hs) {
    __shared__ _Float16 whLds[DIM * DIM];          // 128 KB
    __shared__ _Float16 hLds[2][2][16 * DIM];      // 32 KB (2 M-tiles x 2 bufs)

    const int tid = threadIdx.x;

    // Stage Wh (f32 -> f16); zero-init both h staging buffers (h0 = 0).
    {
        const float4* src = reinterpret_cast<const float4*>(Wh);
        for (int i = tid; i < (DIM * DIM) / 4; i += 128) {
            float4 f = src[i];
            _Float16* d = whLds + i * 4;
            d[0] = (_Float16)f.x; d[1] = (_Float16)f.y;
            d[2] = (_Float16)f.z; d[3] = (_Float16)f.w;
        }
        _Float16* hz = &hLds[0][0][0];
        for (int i = tid; i < 2 * 2 * 16 * DIM; i += 128) hz[i] = (_Float16)0.0f;
    }
    __syncthreads();

    const int wave = tid >> 5;
    const int lane = tid & 31;
    const int half = lane >> 4;
    const int mrow = lane & 15;
    const int pair = wave >> 1;        // local M-tile (0/1)
    const int nh   = wave & 1;         // N half: cols nh*128 .. nh*128+127
    const int row0 = blockIdx.x * 32 + pair * 16;
    const int col0 = nh * 128 + mrow;  // lane's first column

    // Per-lane invariant bases (all hot-loop addressing is constant-offset).
    const _Float16* const whBase = whLds + (size_t)col0 * DIM + half * 8;
    _Float16* const hR0 = &hLds[pair][0][0] + mrow * DIM + half * 8;          // A reads
    _Float16* const hR1 = &hLds[pair][1][0] + mrow * DIM + half * 8;
    _Float16* const hW0 = &hLds[pair][0][0] + (half * 8) * DIM + col0;        // D writes
    _Float16* const hW1 = &hLds[pair][1][0] + (half * 8) * DIM + col0;
    float* hsPtr = hs + (size_t)(row0 + half * 8) * DIM + col0;

    // Bias values for this lane's 8 column tiles, kept in registers.
    float bhv[8];
#pragma unroll
    for (int ntl = 0; ntl < 8; ++ntl) bhv[ntl] = bh[col0 + ntl * 16];

    for (int t = 0; t < T_STEPS; ++t) {
        const _Float16* ar = (t & 1) ? hR1 : hR0;   // read prev h
        _Float16*       aw = (t & 1) ? hW0 : hW1;   // write new h

        // A fragments (16x32 f16) from staging: chunks at +0 and +32 bytes.
        v16h af[8];
#pragma unroll
        for (int kk = 0; kk < 8; ++kk)
            af[kk] = load_frag(ar + kk * 32, ar + kk * 32 + 16);

        const _Float16* bp = whBase;
#pragma unroll
        for (int ntl = 0; ntl < 8; ++ntl) {
            v16h bf[8];
#pragma unroll
            for (int kk = 0; kk < 8; ++kk)
                bf[kk] = load_frag(bp + kk * 32, bp + kk * 32 + 16);

            // Accumulator init = xw_t tile (constant offsets off hsPtr).
            v8f c;
#pragma unroll
            for (int e = 0; e < 8; ++e)
                c[e] = hsPtr[e * DIM + ntl * 16];

#pragma unroll
            for (int kk = 0; kk < 8; ++kk)
                c = __builtin_amdgcn_wmma_f32_16x16x32_f16(
                        false, af[kk], false, bf[kk], (short)0, c, false, false);

            const float bb = bhv[ntl];
#pragma unroll
            for (int e = 0; e < 8; ++e) {
                float v  = c[e] + bb;
                float ex = __expf(2.0f * v);
                float hv = 1.0f - 2.0f / (ex + 1.0f);      // tanh, saturates to +-1
                hsPtr[e * DIM + ntl * 16] = hv;            // hs output (f32)
                aw[e * DIM + ntl * 16] = (_Float16)hv;     // next step's A (f16)
            }
            bp += 16 * DIM;
        }

        __syncthreads();           // staging buffer complete before next reads
        hsPtr += BATCH * DIM;
    }

    // h_final = hs[T-1]; copy from the last slab (values still in L2).
    const float* lastP = hsPtr - BATCH * DIM;
    float* hfPtr = hFinal + (size_t)(row0 + half * 8) * DIM + col0;
#pragma unroll
    for (int ntl = 0; ntl < 8; ++ntl)
#pragma unroll
        for (int e = 0; e < 8; ++e)
            hfPtr[e * DIM + ntl * 16] = lastP[e * DIM + ntl * 16];
}

// ---------------------------------------------------------------------------
extern "C" void kernel_launch(void* const* d_in, const int* in_sizes, int n_in,
                              void* d_out, int out_size, void* d_ws, size_t ws_size,
                              hipStream_t stream) {
    const float* x  = (const float*)d_in[0];   // [T, B, 256]
    const float* Wi = (const float*)d_in[1];   // [256, 256]
    const float* Wh = (const float*)d_in[2];   // [256, 256]
    const float* bh = (const float*)d_in[3];   // [256]

    float* out    = (float*)d_out;
    float* hFinal = out;                       // [B, 256]
    float* hsOut  = out + BATCH * DIM;         // [T, B, 256]

    xwi_gemm<<<256, 128, 0, stream>>>(x, Wi, hsOut);
    rnn_scan<<<4, 128, 0, stream>>>(Wh, bh, hFinal, hsOut);
}
